// FasterRCNN_4028679323773
// MI455X (gfx1250) — compile-verified, matrix operand order checked by reading, not run on hardware
//
#include <hip/hip_runtime.h>
#include <hip/hip_bf16.h>

// ---------------------------------------------------------------------------
// Faster-RCNN forward for gfx1250 (MI455X).
// All matrix work on v_wmma_f32_16x16x32_bf16 (bf16 in, f32 accumulate).
// Activations bf16 HWC. FC GEMM: LDS-staged 32x64 macro-tiles per block.
// Convs 2..5: implicit GEMM, tap-outer / channel-inner loops so the hot loop
// is pure b128 loads + WMMA (no division). Conv1 (Cin=3): explicit im2col.
// ---------------------------------------------------------------------------

typedef __bf16 bf16;
typedef __attribute__((ext_vector_type(4)))  __bf16 bf16x4;
typedef __attribute__((ext_vector_type(8)))  __bf16 bf16x8;
typedef __attribute__((ext_vector_type(16))) __bf16 v16bf;
typedef __attribute__((ext_vector_type(8)))  float  v8f;

#define IMG_SZ   800
#define FEAT_HW  50
#define NCELL    2500
#define NANCH    22500
#define NROIS    300
#define NCLASS   21
#define FC_K     25088

#define CAT16(x0, x1) __builtin_shufflevector(x0, x1, 0,1,2,3,4,5,6,7,8,9,10,11,12,13,14,15)

// ===================== LDS-staged GEMM: C = act(A * B^T) =====================
// A: bf16 [M x K] row-major (K % 32 == 0). B: f32 [N x K] row-major.
// Block = 256 thr = 8 waves = 2x4 grid of 16x16 WMMA tiles (32 x 64 macro).
__global__ __launch_bounds__(256) void gemm_wmma(
    const bf16* __restrict__ A, const float* __restrict__ Bw,
    int M, int N, int K, bf16* __restrict__ Cb, float* __restrict__ Cf, int relu)
{
  __shared__ __align__(16) bf16 sA[32 * 32];   // [row][k]
  __shared__ __align__(16) bf16 sB[64 * 32];   // [col][k]

  const int tid  = threadIdx.x;
  const int lane = tid & 31, wave = tid >> 5;
  const int wr = wave >> 2, wc = wave & 3;          // 2 x 4 wave grid
  const int nTN = (N + 63) / 64;
  const int bm = (blockIdx.x / nTN) * 32;
  const int bn = (blockIdx.x % nTN) * 64;
  const int h = lane >> 4, r = lane & 15;

  // cooperative staging: A = 32 rows x 32 K (4 bf16/thr), B = 64 x 32 (8/thr)
  const int a_row = tid >> 3, a_k = (tid & 7) * 4;
  const int b_col = tid >> 2, b_k = (tid & 3) * 8;
  const int ar = (bm + a_row < M) ? (bm + a_row) : (M - 1);
  const int bc = (bn + b_col < N) ? (bn + b_col) : (N - 1);
  const bf16*  agp = A  + (size_t)ar * K;
  const float* bgp = Bw + (size_t)bc * K;

  v8f acc = {};
  for (int k0 = 0; k0 < K; k0 += 32) {
    if (k0 + 32 < K) {                              // global_prefetch_b8
      __builtin_prefetch(agp + k0 + 32 + a_k, 0, 1);
      __builtin_prefetch(bgp + k0 + 32 + b_k, 0, 1);
    }
    __syncthreads();                                // LDS reuse fence
    *(bf16x4*)(sA + a_row * 32 + a_k) = *(const bf16x4*)(agp + k0 + a_k);
    bf16x8 bcv;
#pragma unroll
    for (int i = 0; i < 8; ++i) bcv[i] = (bf16)bgp[k0 + b_k + i];
    *(bf16x8*)(sB + b_col * 32 + b_k) = bcv;
    __syncthreads();

    const bf16* aR = sA + (wr * 16 + r) * 32;       // fragment row / col
    const bf16* bR = sB + (wc * 16 + r) * 32;
    bf16x8 a0 = *(const bf16x8*)(aR + h * 8);
    bf16x8 a1 = *(const bf16x8*)(aR + 16 + h * 8);
    bf16x8 b0 = *(const bf16x8*)(bR + h * 8);
    bf16x8 b1 = *(const bf16x8*)(bR + 16 + h * 8);
    v16bf av = CAT16(a0, a1);
    v16bf bv = CAT16(b0, b1);
    acc = __builtin_amdgcn_wmma_f32_16x16x32_bf16(false, av, false, bv,
                                                  (short)0, acc, false, false);
  }

  const int tm = bm + wr * 16, tn = bn + wc * 16;
#pragma unroll
  for (int v = 0; v < 8; ++v) {
    const int m = tm + v + h * 8, n = tn + r;
    if (m < M && n < N) {
      float x = acc[v];
      if (relu) x = x > 0.0f ? x : 0.0f;
      if (Cb) Cb[(size_t)m * N + n] = (bf16)x;
      if (Cf) Cf[(size_t)m * N + n] = x;
    }
  }
}

// =============== GEMM variant with bf16 B (conv1 im2col path) ===============
__global__ __launch_bounds__(256) void gemm_wmma_bb(
    const bf16* __restrict__ A, const bf16* __restrict__ Bw,
    int M, int N, int K, bf16* __restrict__ Cb, int relu)
{
  __shared__ __align__(16) bf16 sA[32 * 32];
  __shared__ __align__(16) bf16 sB[64 * 32];
  const int tid  = threadIdx.x;
  const int lane = tid & 31, wave = tid >> 5;
  const int wr = wave >> 2, wc = wave & 3;
  const int nTN = (N + 63) / 64;
  const int bm = (blockIdx.x / nTN) * 32;
  const int bn = (blockIdx.x % nTN) * 64;
  const int h = lane >> 4, r = lane & 15;
  const int a_row = tid >> 3, a_k = (tid & 7) * 4;
  const int b_col = tid >> 2, b_k = (tid & 3) * 8;
  const int ar = (bm + a_row < M) ? (bm + a_row) : (M - 1);
  const int bc = (bn + b_col < N) ? (bn + b_col) : (N - 1);
  const bf16* agp = A  + (size_t)ar * K;
  const bf16* bgp = Bw + (size_t)bc * K;

  v8f acc = {};
  for (int k0 = 0; k0 < K; k0 += 32) {
    __syncthreads();
    *(bf16x4*)(sA + a_row * 32 + a_k) = *(const bf16x4*)(agp + k0 + a_k);
    *(bf16x8*)(sB + b_col * 32 + b_k) = *(const bf16x8*)(bgp + k0 + b_k);
    __syncthreads();
    const bf16* aR = sA + (wr * 16 + r) * 32;
    const bf16* bR = sB + (wc * 16 + r) * 32;
    bf16x8 a0 = *(const bf16x8*)(aR + h * 8);
    bf16x8 a1 = *(const bf16x8*)(aR + 16 + h * 8);
    bf16x8 b0 = *(const bf16x8*)(bR + h * 8);
    bf16x8 b1 = *(const bf16x8*)(bR + 16 + h * 8);
    v16bf av = CAT16(a0, a1);
    v16bf bv = CAT16(b0, b1);
    acc = __builtin_amdgcn_wmma_f32_16x16x32_bf16(false, av, false, bv,
                                                  (short)0, acc, false, false);
  }
  const int tm = bm + wr * 16, tn = bn + wc * 16;
#pragma unroll
  for (int v = 0; v < 8; ++v) {
    const int m = tm + v + h * 8, n = tn + r;
    if (m < M && n < N) {
      float x = acc[v];
      if (relu) x = x > 0.0f ? x : 0.0f;
      Cb[(size_t)m * N + n] = (bf16)x;
    }
  }
}

// =============== implicit-GEMM 3x3 conv (Cin % 32 == 0, HWC) ================
// wt: bf16 [Cout][K], K = 9*Cin, k = (ky*3+kx)*Cin + ci. One wave = 16x16 tile.
// Tap-outer loop: all index math / bounds predicate hoisted out of the
// channel-slab inner loop, which is pure b128 loads + WMMA.
__global__ __launch_bounds__(256) void conv_wmma(
    const bf16* __restrict__ in, int Cin, int H, int W,
    const bf16* __restrict__ wt, bf16* __restrict__ out,
    int OH, int OW, int Cout, int stride, int pad)
{
  const int M = OH * OW, N = Cout, K = 9 * Cin;
  const int lane = threadIdx.x & 31, wave = threadIdx.x >> 5;
  const int tN = N >> 4, tM = (M + 15) >> 4;
  const int tile = blockIdx.x * 8 + wave;
  if (tile >= tM * tN) return;
  const int tm = (tile / tN) << 4, tn = (tile % tN) << 4;
  const int h = lane >> 4, r = lane & 15;
  const int mrow = (tm + r < M) ? (tm + r) : (M - 1);   // clamp; store-guarded
  const int ncol = tn + r;
  const int oy = mrow / OW, ox = mrow % OW;
  const int lo = h * 8;                                  // lane K sub-offset
  const bf16* wrow = wt + (size_t)ncol * K + lo;

  v8f acc = {};
#pragma unroll 3
  for (int j = 0; j < 9; ++j) {                          // filter tap (ky,kx)
    const int ky = j / 3, kx = j % 3;                    // const divisors
    const int iy = oy * stride + ky - pad;
    const int ix = ox * stride + kx - pad;
    const bool inb = (iy >= 0 && iy < H && ix >= 0 && ix < W);
    const bf16* ap = in + ((size_t)iy * W + ix) * Cin + lo;
    const bf16* wp = wrow + j * Cin;
    for (int c0 = 0; c0 < Cin; c0 += 32) {               // channel slabs
      bf16x8 a0 = {}, a1 = {};
      if (inb) {
        a0 = *(const bf16x8*)(ap + c0);
        a1 = *(const bf16x8*)(ap + c0 + 16);
      }
      bf16x8 b0 = *(const bf16x8*)(wp + c0);
      bf16x8 b1 = *(const bf16x8*)(wp + c0 + 16);
      v16bf av = CAT16(a0, a1);
      v16bf bv = CAT16(b0, b1);
      acc = __builtin_amdgcn_wmma_f32_16x16x32_bf16(false, av, false, bv,
                                                    (short)0, acc, false, false);
    }
  }
#pragma unroll
  for (int v = 0; v < 8; ++v) {
    const int m = tm + v + h * 8, n = tn + r;
    if (m < M) {
      float x = acc[v];
      x = x > 0.0f ? x : 0.0f;
      out[(size_t)m * N + n] = (bf16)x;
    }
  }
}

// -------------------------- weight prep kernels -----------------------------
__global__ void wpad1(const float* __restrict__ w, bf16* __restrict__ wt) {
  int idx = blockIdx.x * blockDim.x + threadIdx.x;      // 64 x 32
  if (idx >= 64 * 32) return;
  int n = idx >> 5, k = idx & 31;                       // k = ci*9 + j (OIHW flat)
  wt[idx] = (k < 27) ? (bf16)w[n * 27 + k] : (bf16)0.0f;
}
__global__ void wtrans(const float* __restrict__ w, bf16* __restrict__ wt,
                       int Cout, int Cin) {
  int K = 9 * Cin;
  int idx = blockIdx.x * blockDim.x + threadIdx.x;
  if (idx >= Cout * K) return;
  int n = idx / K, k = idx % K;
  int j = k / Cin, ci = k % Cin;                        // k = j*Cin + ci
  wt[idx] = (bf16)w[((size_t)n * Cin + ci) * 9 + j];
}

// -------------------- conv1 im2col: [160000 x 32] bf16 ----------------------
__global__ void im2col1(const float* __restrict__ x, bf16* __restrict__ col) {
  int idx = blockIdx.x * blockDim.x + threadIdx.x;
  if (idx >= 160000 * 32) return;
  int m = idx >> 5, k = idx & 31;
  bf16 v = (bf16)0.0f;
  if (k < 27) {
    int ci = k / 9, j = k % 9, ky = j / 3, kx = j % 3;
    int oy = m / 400, ox = m % 400;
    int iy = oy * 2 + ky, ix = ox * 2 + kx;             // pad_before = 0
    if (iy < 800 && ix < 800) v = (bf16)x[((size_t)ci * 800 + iy) * 800 + ix];
  }
  col[idx] = v;
}

// ---------------- anchors + loc2bbox decode + fg softmax --------------------
__global__ void decode_kernel(const float* __restrict__ locs,
                              const float* __restrict__ scores,
                              float* __restrict__ fg, float* __restrict__ boxes)
{
  int i = blockIdx.x * blockDim.x + threadIdx.x;
  if (i >= NANCH) return;
  const int m = i / 9, a = i % 9;
  const int iy = m / FEAT_HW, ix = m % FEAT_HW;
  const int ri = a / 3, si = a % 3;
  const float scales[3] = {8.0f, 16.0f, 32.0f};
  const float ratios[3] = {0.5f, 1.0f, 2.0f};
  const float base = 16.0f * scales[si];
  const float sq = sqrtf(ratios[ri]);
  const float ha = sq * base, wa = base / sq;
  const float cy = (iy + 0.5f) * 16.0f, cx = (ix + 0.5f) * 16.0f;

  const float dy = locs[i * 4 + 0], dx = locs[i * 4 + 1];
  const float dh = locs[i * 4 + 2], dw = locs[i * 4 + 3];
  const float ncy = dy * ha + cy, ncx = dx * wa + cx;
  const float nh = expf(dh) * ha, nw = expf(dw) * wa;
  float b0 = ncy - 0.5f * nh, b1 = ncx - 0.5f * nw;
  float b2 = ncy + 0.5f * nh, b3 = ncx + 0.5f * nw;
  boxes[i * 4 + 0] = fminf(fmaxf(b0, 0.0f), (float)IMG_SZ);
  boxes[i * 4 + 1] = fminf(fmaxf(b1, 0.0f), (float)IMG_SZ);
  boxes[i * 4 + 2] = fminf(fmaxf(b2, 0.0f), (float)IMG_SZ);
  boxes[i * 4 + 3] = fminf(fmaxf(b3, 0.0f), (float)IMG_SZ);

  const float s0 = scores[m * 18 + a * 2 + 0];
  const float s1 = scores[m * 18 + a * 2 + 1];
  fg[i] = 1.0f / (1.0f + expf(s0 - s1));
}

// ------------- exact top-k(300) via O(N^2) ranking (deterministic) ----------
__global__ void topk_kernel(const float* __restrict__ fg,
                            const float* __restrict__ boxes,
                            float* __restrict__ rois_ws,
                            float* __restrict__ out_rois,
                            int*   __restrict__ out_idx)
{
  int i = blockIdx.x * blockDim.x + threadIdx.x;
  if (i >= NANCH) return;
  const float v = fg[i];
  int rank = 0;
  for (int j = 0; j < NANCH; ++j) {
    const float u = fg[j];
    rank += (u > v) || (u == v && j < i);   // lax.top_k tie order
  }
  if (rank < NROIS) {
#pragma unroll
    for (int d = 0; d < 4; ++d) {
      const float bd = boxes[i * 4 + d];
      rois_ws[rank * 4 + d]  = bd;
      out_rois[rank * 4 + d] = bd;
    }
  }
  if (i < NROIS) out_idx[i] = 0;
}

// ------------------------------- ROI pooling --------------------------------
__global__ void roipool_kernel(const bf16* __restrict__ feat,
                               const float* __restrict__ rois,
                               bf16* __restrict__ pooled)
{
  int idx = blockIdx.x * blockDim.x + threadIdx.x;
  if (idx >= NROIS * 512 * 49) return;
  const int m  = idx / (512 * 49);
  const int re = idx % (512 * 49);
  const int c  = re / 49;
  const int pq = re % 49;
  const int py = pq / 7, px = pq % 7;
  const float* rr = rois + m * 4;
  const float ys = rr[0] + (rr[2] - rr[0]) * ((py + 0.5f) / 7.0f);
  const float xs = rr[1] + (rr[3] - rr[1]) * ((px + 0.5f) / 7.0f);
  int yi = (int)(ys * (1.0f / 16.0f));
  int xi = (int)(xs * (1.0f / 16.0f));
  yi = yi < 0 ? 0 : (yi > FEAT_HW - 1 ? FEAT_HW - 1 : yi);
  xi = xi < 0 ? 0 : (xi > FEAT_HW - 1 ? FEAT_HW - 1 : xi);
  pooled[(size_t)m * FC_K + c * 49 + pq] = feat[((size_t)yi * FEAT_HW + xi) * 512 + c];
}

// ----------------------------------------------------------------------------
extern "C" void kernel_launch(void* const* d_in, const int* in_sizes, int n_in,
                              void* d_out, int out_size, void* d_ws, size_t ws_size,
                              hipStream_t stream) {
  (void)in_sizes; (void)n_in; (void)out_size; (void)ws_size;
  const float* x        = (const float*)d_in[0];
  const float* w1       = (const float*)d_in[1];
  const float* w2       = (const float*)d_in[2];
  const float* w3       = (const float*)d_in[3];
  const float* w4       = (const float*)d_in[4];
  const float* rpn_w    = (const float*)d_in[5];
  const float* loc_w    = (const float*)d_in[6];
  const float* score_w  = (const float*)d_in[7];
  const float* fc1_w    = (const float*)d_in[8];
  const float* fc2_w    = (const float*)d_in[9];
  const float* clsloc_w = (const float*)d_in[10];
  const float* scfc_w   = (const float*)d_in[11];
  float* out = (float*)d_out;

  char* p = (char*)d_ws;
  auto alloc = [&](size_t bytes) -> char* {
    char* q = p; p += (bytes + 255) & ~(size_t)255; return q;
  };
  bf16*  act1   = (bf16*)alloc((size_t)400 * 400 * 64  * sizeof(bf16));
  bf16*  act2   = (bf16*)alloc((size_t)200 * 200 * 128 * sizeof(bf16));
  bf16*  act3   = (bf16*)alloc((size_t)100 * 100 * 256 * sizeof(bf16));
  bf16*  act4   = (bf16*)alloc((size_t)50  * 50  * 512 * sizeof(bf16));
  bf16*  rpnh   = (bf16*)alloc((size_t)50  * 50  * 512 * sizeof(bf16));
  bf16*  pooled = (bf16*)alloc((size_t)NROIS * FC_K    * sizeof(bf16));
  bf16*  h1     = (bf16*)alloc((size_t)NROIS * 4096    * sizeof(bf16));
  bf16*  h2     = (bf16*)alloc((size_t)NROIS * 4096    * sizeof(bf16));
  bf16*  col1   = (bf16*)alloc((size_t)160000 * 32     * sizeof(bf16));
  bf16*  w1t    = (bf16*)alloc((size_t)64  * 32        * sizeof(bf16));
  bf16*  w2t    = (bf16*)alloc((size_t)128 * 576       * sizeof(bf16));
  bf16*  w3t    = (bf16*)alloc((size_t)256 * 1152      * sizeof(bf16));
  bf16*  w4t    = (bf16*)alloc((size_t)512 * 2304      * sizeof(bf16));
  bf16*  rpnt   = (bf16*)alloc((size_t)512 * 4608      * sizeof(bf16));
  float* locs   = (float*)alloc((size_t)NANCH * 4  * sizeof(float));
  float* scores = (float*)alloc((size_t)NCELL * 18 * sizeof(float));
  float* fg     = (float*)alloc((size_t)NANCH * sizeof(float));
  float* boxes  = (float*)alloc((size_t)NANCH * 4 * sizeof(float));
  float* rois   = (float*)alloc((size_t)NROIS * 4 * sizeof(float));

  auto gemmb = [](int M, int N) { return ((M + 31) / 32) * ((N + 63) / 64); };
  auto convb = [](int M, int N) {
    return ((((M + 15) / 16) * (N / 16)) + 7) / 8;
  };

  // ---- weight prep + conv1 im2col ----
  wpad1<<<(64 * 32 + 255) / 256, 256, 0, stream>>>(w1, w1t);
  wtrans<<<(128 * 576  + 255) / 256, 256, 0, stream>>>(w2, w2t, 128, 64);
  wtrans<<<(256 * 1152 + 255) / 256, 256, 0, stream>>>(w3, w3t, 256, 128);
  wtrans<<<(512 * 2304 + 255) / 256, 256, 0, stream>>>(w4, w4t, 512, 256);
  wtrans<<<(512 * 4608 + 255) / 256, 256, 0, stream>>>(rpn_w, rpnt, 512, 512);
  im2col1<<<(160000 * 32 + 255) / 256, 256, 0, stream>>>(x, col1);

  // ---- conv1 as GEMM on im2col (bf16 A, bf16 padded B) ----
  gemm_wmma_bb<<<gemmb(160000, 64), 256, 0, stream>>>(
      col1, w1t, 160000, 64, 32, act1, 1);

  // ---- backbone convs 2..5 ----
  conv_wmma<<<convb(40000, 128), 256, 0, stream>>>(
      act1, 64, 400, 400, w2t, act2, 200, 200, 128, 2, 0);
  conv_wmma<<<convb(10000, 256), 256, 0, stream>>>(
      act2, 128, 200, 200, w3t, act3, 100, 100, 256, 2, 0);
  conv_wmma<<<convb(2500, 512), 256, 0, stream>>>(
      act3, 256, 100, 100, w4t, act4, 50, 50, 512, 2, 0);
  conv_wmma<<<convb(2500, 512), 256, 0, stream>>>(
      act4, 512, 50, 50, rpnt, rpnh, 50, 50, 512, 1, 1);

  // ---- RPN 1x1 heads as GEMM (A = rpnh [2500 x 512] bf16 HWC) ----
  gemm_wmma<<<gemmb(NCELL, 36), 256, 0, stream>>>(
      rpnh, loc_w, NCELL, 36, 512, nullptr, locs, 0);
  gemm_wmma<<<gemmb(NCELL, 18), 256, 0, stream>>>(
      rpnh, score_w, NCELL, 18, 512, nullptr, scores, 0);

  // ---- decode + top-k + ROI pooling ----
  decode_kernel<<<(NANCH + 255) / 256, 256, 0, stream>>>(locs, scores, fg, boxes);
  float* out_rois = out + NROIS * (NCLASS * 4) + NROIS * NCLASS;
  int*   out_idx  = (int*)(out_rois + NROIS * 4);
  topk_kernel<<<(NANCH + 255) / 256, 256, 0, stream>>>(fg, boxes, rois, out_rois, out_idx);
  roipool_kernel<<<(NROIS * 512 * 49 + 255) / 256, 256, 0, stream>>>(act4, rois, pooled);

  // ---- FC head ----
  gemm_wmma<<<gemmb(NROIS, 4096), 256, 0, stream>>>(
      pooled, fc1_w, NROIS, 4096, FC_K, h1, nullptr, 1);
  gemm_wmma<<<gemmb(NROIS, 4096), 256, 0, stream>>>(
      h1, fc2_w, NROIS, 4096, 4096, h2, nullptr, 1);
  gemm_wmma<<<gemmb(NROIS, NCLASS * 4), 256, 0, stream>>>(
      h2, clsloc_w, NROIS, NCLASS * 4, 4096, nullptr, out, 0);
  gemm_wmma<<<gemmb(NROIS, NCLASS), 256, 0, stream>>>(
      h2, scfc_w, NROIS, NCLASS, 4096, nullptr, out + NROIS * (NCLASS * 4), 0);
}
